// ved_TP_43319040147931
// MI455X (gfx1250) — compile-verified, matrix operand order checked by reading, not run on hardware
//
#include <hip/hip_runtime.h>

// ---------------------------------------------------------------------------
// CDNA5 (gfx1250) tensor-parallel attention forward.
//  - All matmuls: v_wmma_f32_16x16x32_bf16 (wave32 WMMA, f32 accumulate)
//  - Main GEMMs: double-buffered GLOBAL_LOAD_ASYNC_TO_LDS_B128 staging
//    (ASYNCcnt-tracked, overlapped with WMMA compute)
// ---------------------------------------------------------------------------

typedef __attribute__((ext_vector_type(16))) __bf16 v16bf;
typedef __attribute__((ext_vector_type(8)))  float  v8f;

#define BM 128
#define BN 128
#define BK 32
#define LDSS 40   // BK + 8 halves padding to de-conflict LDS banks

// Problem constants
#define DMODEL 4096
#define DLOCAL 512
#define BB     4
#define SS     2048

__device__ __forceinline__ v16bf ld_frag(const __bf16* p) {
  union { uint4 u[2]; v16bf v; } t;
  t.u[0] = *(const uint4*)(p);
  t.u[1] = *(const uint4*)(p + 8);
  return t.v;
}

// 32B global -> LDS, asynchronous (2 x b128). INST_OFFSET applies to both the
// global and the LDS address, so one address pair covers both halves.
__device__ __forceinline__ void async_copy_b32B(uint32_t lds_off, const __bf16* g) {
  asm volatile("global_load_async_to_lds_b128 %0, %1, off\n\t"
               "global_load_async_to_lds_b128 %0, %1, off offset:16"
               :: "v"(lds_off), "v"((unsigned long long)g)
               : "memory");
}

__global__ __launch_bounds__(256)
void cast_f32_bf16(const float* __restrict__ src, __bf16* __restrict__ dst, int n) {
  int i = blockIdx.x * 256 + threadIdx.x;
  if (i < n) dst[i] = (__bf16)src[i];
}

// ---------------------------------------------------------------------------
// out[M,N] = (A @ Bm^T + bias) * scale
//   A  stored [M,K] row-major (stride lda), Bm stored [N,K] row-major (ldb).
//   128x128 block tile, 8 waves (4x2), each wave = 32x64 via 8 WMMAs/K-step.
//   Async double-buffered LDS staging.
// ---------------------------------------------------------------------------
template <bool OUT_F32>
__global__ __launch_bounds__(256)
void gemm_async(const __bf16* __restrict__ A,
                const __bf16* __restrict__ Bm,
                const float*  __restrict__ bias,
                void* __restrict__ outp,
                int M, int N, int K, int lda, int ldb, float scale)
{
  __shared__ __align__(16) __bf16 As[2][BM * LDSS];
  __shared__ __align__(16) __bf16 Bs[2][BN * LDSS];

  const int tid     = threadIdx.x;
  const int lane    = tid & 31;
  const int wave    = tid >> 5;
  const int waveRow = wave & 3;   // 4 wave-rows of 32
  const int waveCol = wave >> 2;  // 2 wave-cols of 64
  const int blockM  = blockIdx.y * BM;
  const int blockN  = blockIdx.x * BN;

  // staging coordinates: each thread owns one 32B half-row in A and B tiles
  const int sr = tid >> 1;        // 0..127
  const int sh = tid & 1;         // K-half of 16 halves
  const __bf16* gA = A  + (size_t)(blockM + sr) * lda + sh * 16;
  const __bf16* gB = Bm + (size_t)(blockN + sr) * ldb + sh * 16;
  const int sOff = sr * LDSS + sh * 16;

  v8f acc[2][4] = {};

  const int fr = lane & 15;       // fragment row within a 16x32 tile
  const int fh = lane >> 4;       // K-half select

  const int nsteps = K / BK;

  // prologue: stage step 0 into buffer 0
  async_copy_b32B((uint32_t)(size_t)&As[0][sOff], gA);
  async_copy_b32B((uint32_t)(size_t)&Bs[0][sOff], gB);

  for (int t = 0; t < nsteps; ++t) {
    const int cur = t & 1;
    if (t + 1 < nsteps) {
      const int koff = (t + 1) * BK;
      async_copy_b32B((uint32_t)(size_t)&As[cur ^ 1][sOff], gA + koff);
      async_copy_b32B((uint32_t)(size_t)&Bs[cur ^ 1][sOff], gB + koff);
      asm volatile("s_wait_asynccnt 0x4" ::: "memory");  // step-t ops retired
    } else {
      asm volatile("s_wait_asynccnt 0x0" ::: "memory");
    }
    __syncthreads();  // all waves' step-t tiles visible in LDS

    v16bf afrag[2], bfrag[4];
    #pragma unroll
    for (int mt = 0; mt < 2; ++mt)
      afrag[mt] = ld_frag(&As[cur][(waveRow * 32 + mt * 16 + fr) * LDSS + fh * 16]);
    #pragma unroll
    for (int nt = 0; nt < 4; ++nt)
      bfrag[nt] = ld_frag(&Bs[cur][(waveCol * 64 + nt * 16 + fr) * LDSS + fh * 16]);

    #pragma unroll
    for (int mt = 0; mt < 2; ++mt)
      #pragma unroll
      for (int nt = 0; nt < 4; ++nt)
        acc[mt][nt] = __builtin_amdgcn_wmma_f32_16x16x32_bf16(
            false, afrag[mt], false, bfrag[nt], (short)0, acc[mt][nt],
            false, false);

    __syncthreads();  // buffer `cur` free for step t+2's async writes
  }

  // epilogue: C layout = VGPR j -> row j + 8*(lane/16), col = lane%16
  const int ccol = lane & 15;
  const int crow = (lane >> 4) * 8;
  #pragma unroll
  for (int mt = 0; mt < 2; ++mt) {
    #pragma unroll
    for (int nt = 0; nt < 4; ++nt) {
      int colg = blockN + waveCol * 64 + nt * 16 + ccol;
      float bv = bias ? bias[colg] : 0.0f;
      #pragma unroll
      for (int j = 0; j < 8; ++j) {
        int rowg = blockM + waveRow * 32 + mt * 16 + crow + j;
        float v = (acc[mt][nt][j] + bv) * scale;
        if (OUT_F32) ((float*)outp)[(size_t)rowg * N + colg] = v;
        else         ((__bf16*)outp)[(size_t)rowg * N + colg] = (__bf16)v;
      }
    }
  }
}

// ---------------------------------------------------------------------------
// attn^T GEMM: out[M=512,N=2048](bf16) = (V^T) @ (probs^T)
//   A'[m,k] = Vb[k*lda + m]   (Vb is [K x M]; transposed while staging to LDS)
//   Bm = probs [N,K] row-major (rows of probs are columns of probs^T).
// ---------------------------------------------------------------------------
__global__ __launch_bounds__(256)
void gemm_tA(const __bf16* __restrict__ A,
             const __bf16* __restrict__ Bm,
             __bf16* __restrict__ outp,
             int M, int N, int K, int lda, int ldb)
{
  __shared__ __align__(16) __bf16 As[BM * LDSS];
  __shared__ __align__(16) __bf16 Bs[BN * LDSS];

  const int tid     = threadIdx.x;
  const int lane    = tid & 31;
  const int wave    = tid >> 5;
  const int waveRow = wave & 3;
  const int waveCol = wave >> 2;
  const int blockM  = blockIdx.y * BM;
  const int blockN  = blockIdx.x * BN;

  v8f acc[2][4] = {};

  const int fr = lane & 15;
  const int fh = lane >> 4;

  for (int k0 = 0; k0 < K; k0 += BK) {
    {   // A is [K x M]; read coalesced rows, write transposed into LDS
      int kk = tid >> 3;                 // 0..31
      int e0 = (tid & 7) * 16;           // 0..112
      const __bf16* gp = A + (size_t)(k0 + kk) * lda + blockM + e0;
      #pragma unroll
      for (int i = 0; i < 16; ++i)
        As[(e0 + i) * LDSS + kk] = gp[i];
    }
    {   // B tile: Bs[n][k], 128x32 bf16
      int r = tid >> 1;                  // 0..127
      int h = tid & 1;
      const __bf16* gp = Bm + (size_t)(blockN + r) * ldb + k0 + h * 16;
      *(uint4*)&Bs[r * LDSS + h * 16]     = *(const uint4*)gp;
      *(uint4*)&Bs[r * LDSS + h * 16 + 8] = *(const uint4*)(gp + 8);
    }
    __syncthreads();

    v16bf afrag[2], bfrag[4];
    #pragma unroll
    for (int mt = 0; mt < 2; ++mt)
      afrag[mt] = ld_frag(&As[(waveRow * 32 + mt * 16 + fr) * LDSS + fh * 16]);
    #pragma unroll
    for (int nt = 0; nt < 4; ++nt)
      bfrag[nt] = ld_frag(&Bs[(waveCol * 64 + nt * 16 + fr) * LDSS + fh * 16]);

    #pragma unroll
    for (int mt = 0; mt < 2; ++mt)
      #pragma unroll
      for (int nt = 0; nt < 4; ++nt)
        acc[mt][nt] = __builtin_amdgcn_wmma_f32_16x16x32_bf16(
            false, afrag[mt], false, bfrag[nt], (short)0, acc[mt][nt],
            false, false);

    __syncthreads();
  }

  const int ccol = lane & 15;
  const int crow = (lane >> 4) * 8;
  #pragma unroll
  for (int mt = 0; mt < 2; ++mt) {
    #pragma unroll
    for (int nt = 0; nt < 4; ++nt) {
      int colg = blockN + waveCol * 64 + nt * 16 + ccol;
      #pragma unroll
      for (int j = 0; j < 8; ++j) {
        int rowg = blockM + waveRow * 32 + mt * 16 + crow + j;
        outp[(size_t)rowg * N + colg] = (__bf16)acc[mt][nt][j];
      }
    }
  }
}

// Row softmax over 2048 columns (one 256-thread block per row), masked.
__global__ __launch_bounds__(256)
void softmax_rows(const float* __restrict__ scores, const int* __restrict__ mask,
                  __bf16* __restrict__ probs)
{
  const int row = blockIdx.x;
  const int tid = threadIdx.x;
  const float* srow = scores + (size_t)row * SS;
  __bf16* prow = probs + (size_t)row * SS;

  float vals[8];
  float m = -3.0e38f;
  #pragma unroll
  for (int i = 0; i < 8; ++i) {
    int c = tid + i * 256;
    float s = srow[c];
    if (mask[c] == 0) s = -1.0e9f;
    vals[i] = s;
    m = fmaxf(m, s);
  }
  for (int off = 16; off > 0; off >>= 1) m = fmaxf(m, __shfl_xor(m, off, 32));
  __shared__ float red[8];
  if ((tid & 31) == 0) red[tid >> 5] = m;
  __syncthreads();
  m = red[0];
  #pragma unroll
  for (int w = 1; w < 8; ++w) m = fmaxf(m, red[w]);
  __syncthreads();

  float sum = 0.0f;
  #pragma unroll
  for (int i = 0; i < 8; ++i) { vals[i] = __expf(vals[i] - m); sum += vals[i]; }
  for (int off = 16; off > 0; off >>= 1) sum += __shfl_xor(sum, off, 32);
  if ((tid & 31) == 0) red[tid >> 5] = sum;
  __syncthreads();
  sum = 0.0f;
  #pragma unroll
  for (int w = 0; w < 8; ++w) sum += red[w];
  const float inv = 1.0f / sum;
  #pragma unroll
  for (int i = 0; i < 8; ++i) prow[tid + i * 256] = (__bf16)(vals[i] * inv);
}

extern "C" void kernel_launch(void* const* d_in, const int* in_sizes, int n_in,
                              void* d_out, int out_size, void* d_ws, size_t ws_size,
                              hipStream_t stream) {
  (void)in_sizes; (void)n_in; (void)out_size; (void)ws_size;
  const float* x    = (const float*)d_in[0];
  const int*   mask = (const int*)d_in[1];
  const float* Wq = (const float*)d_in[2];
  const float* bq = (const float*)d_in[3];
  const float* Wk = (const float*)d_in[4];
  const float* bk = (const float*)d_in[5];
  const float* Wv = (const float*)d_in[6];
  const float* bv = (const float*)d_in[7];
  const float* Wo = (const float*)d_in[8];
  const float* bo = (const float*)d_in[9];
  float* out = (float*)d_out;

  // Workspace layout (104 MB). scores/probs/attnT overlay the x_bf16 region,
  // which is dead once Q/K/V are produced.
  char* ws = (char*)d_ws;
  __bf16* xbf    = (__bf16*)(ws);                              // 64 MB
  float*  scores = (float*)(ws);                               // 16 MB (reuse)
  __bf16* probs  = (__bf16*)(ws + (size_t)(16u << 20));        //  8 MB (reuse)
  __bf16* attnT  = (__bf16*)(ws + (size_t)(24u << 20));        //  8 MB (reuse)
  __bf16* wqb    = (__bf16*)(ws + (size_t)(64u << 20));
  __bf16* wkb    = wqb + (size_t)DLOCAL * DMODEL;
  __bf16* wvb    = wkb + (size_t)DLOCAL * DMODEL;
  __bf16* wob    = wvb + (size_t)DLOCAL * DMODEL;
  __bf16* Q      = (__bf16*)(ws + (size_t)(80u << 20));
  __bf16* Kt     = Q  + (size_t)BB * SS * DLOCAL;
  __bf16* V      = Kt + (size_t)BB * SS * DLOCAL;

  const int nX = BB * SS * DMODEL;        // 33,554,432
  const int nW = DLOCAL * DMODEL;         //  2,097,152

  cast_f32_bf16<<<(nX + 255) / 256, 256, 0, stream>>>(x,  xbf, nX);
  cast_f32_bf16<<<(nW + 255) / 256, 256, 0, stream>>>(Wq, wqb, nW);
  cast_f32_bf16<<<(nW + 255) / 256, 256, 0, stream>>>(Wk, wkb, nW);
  cast_f32_bf16<<<(nW + 255) / 256, 256, 0, stream>>>(Wv, wvb, nW);
  cast_f32_bf16<<<(nW + 255) / 256, 256, 0, stream>>>(Wo, wob, nW);

  const float qscale = 0.08838834764831845f;  // 1/sqrt(d_k=128), folded into Q
  const int Mx = BB * SS;                     // 8192

  dim3 gProj(DLOCAL / BN, Mx / BM);           // (4, 64)
  gemm_async<false><<<gProj, 256, 0, stream>>>(
      xbf, wqb, bq, Q,  Mx, DLOCAL, DMODEL, DMODEL, DMODEL, qscale);
  gemm_async<false><<<gProj, 256, 0, stream>>>(
      xbf, wkb, bk, Kt, Mx, DLOCAL, DMODEL, DMODEL, DMODEL, 1.0f);
  gemm_async<false><<<gProj, 256, 0, stream>>>(
      xbf, wvb, bv, V,  Mx, DLOCAL, DMODEL, DMODEL, DMODEL, 1.0f);

  for (int b = 0; b < BB; ++b) {
    const __bf16* Qb = Q  + (size_t)b * SS * DLOCAL;
    const __bf16* Kb = Kt + (size_t)b * SS * DLOCAL;
    const __bf16* Vb = V  + (size_t)b * SS * DLOCAL;
    __bf16* aTb = attnT + (size_t)b * DLOCAL * SS;  // [512, 2048] row-major

    // scores[q,k] = Qb @ Kb^T   (scale already folded into Q)
    dim3 gS(SS / BN, SS / BM);                 // (16, 16)
    gemm_async<true><<<gS, 256, 0, stream>>>(
        Qb, Kb, nullptr, scores, SS, SS, DLOCAL, DLOCAL, DLOCAL, 1.0f);

    softmax_rows<<<SS, 256, 0, stream>>>(scores, mask + (size_t)b * SS, probs);

    // attnT[e,q] = sum_k V[k,e] * probs[q,k]  == (probs @ V)^T
    dim3 gA(SS / BN, DLOCAL / BM);             // (16, 4)
    gemm_tA<<<gA, 256, 0, stream>>>(
        Vb, probs, aTb, DLOCAL, SS, SS, DLOCAL, SS);
  }

  // attnT buffer, reinterpreted as [8192, 512] row-major, is exactly the
  // reference's swapaxes(1,2).reshape(B,S,d_local) permutation.
  dim3 gO(DMODEL / BN, Mx / BM);               // (32, 64)
  gemm_async<true><<<gO, 256, 0, stream>>>(
      attnT, wob, bo, out, Mx, DMODEL, DLOCAL, DLOCAL, DLOCAL, 1.0f);
}
